// NeRF_7258494730406
// MI455X (gfx1250) — matrix operand order repaired and checked by reading, not compiled, not run on hardware
//
#include <hip/hip_runtime.h>
#include <hip/hip_bf16.h>
#include <math.h>

// ---------- CDNA5 WMMA types ----------
typedef __attribute__((ext_vector_type(16))) _Float16 v16h;
typedef __attribute__((ext_vector_type(8)))  float    v8f;
typedef __attribute__((ext_vector_type(4)))  unsigned int u32x4;

union ABfrag { v16h v; u32x4 u[2]; };

// All weight pointers (pre-packed f16, transposed [N][Kpad]) + fp32 biases/heads.
struct MlpWeights {
  const _Float16 *W0T,*W1T,*W2T,*W3T,*W4T,*W5T,*W6T,*W7T,*featT,*branchT;
  const float *b0,*b1,*b2,*b3,*b4,*b5,*b6,*b7,*featb,*branchb;
  const float *sigw,*sigb,*rgbw,*rgbb;
};

#define MT 64           // points per block (4 M-tiles of 16)
#define LDA 256         // Xa row stride (halfs)
#define LDB 320         // Xb row stride (halfs): 256 + 63 skip + pad

// ---------- pack fp32 W[K][N] -> f16 WT[N][Kpad] (zero-pad K) ----------
__global__ void pack_wt_kernel(const float* __restrict__ W, _Float16* __restrict__ WT,
                               int K, int N, int Kpad) {
  int i = blockIdx.x * 256 + threadIdx.x;
  int total = N * Kpad;
  if (i >= total) return;
  int n = i / Kpad, k = i - n * Kpad;
  WT[i] = (k < K) ? (_Float16)W[(size_t)k * N + n] : (_Float16)0.f;
}

// ---------- positional encoding: [x(3), sin(30), cos(30), pad0] ----------
__device__ __forceinline__ void write_enc(_Float16* row, float x, float y, float z) {
  row[0] = (_Float16)x; row[1] = (_Float16)y; row[2] = (_Float16)z;
  float v[3] = {x, y, z};
#pragma unroll
  for (int c = 0; c < 3; ++c) {
    float f = 1.f;
#pragma unroll
    for (int j = 0; j < 10; ++j) {
      float a = v[c] * f;
      row[3  + c*10 + j] = (_Float16)__sinf(a);
      row[33 + c*10 + j] = (_Float16)__cosf(a);
      f *= 2.f;
    }
  }
  row[63] = (_Float16)0.f;
}

// ---------- WMMA GEMM layer ----------
// Y[MT][N] = act(X[MT][Kpad] @ WT^T + b),  N = NTW*128.
// Wave w owns N-tiles {w, w+8, ...}; per k-step it loads 4 A fragments once and
// reuses them against each of its B fragments (4 WMMAs per B load).
template<int NTW>
__device__ __forceinline__ void gemm_layer(const _Float16* X, int ldx, int Kpad,
    const _Float16* __restrict__ WT, const float* __restrict__ bias,
    _Float16* Y, int ldy, bool do_relu)
{
  int tid  = threadIdx.x;
  int lane = tid & 31, wave = tid >> 5;
  int lh = lane & 15, lg = lane >> 4;          // half-wave group

  v8f acc[NTW][4];
#pragma unroll
  for (int j = 0; j < NTW; ++j)
#pragma unroll
    for (int mt = 0; mt < 4; ++mt)
      acc[j][mt] = (v8f){0.f,0.f,0.f,0.f,0.f,0.f,0.f,0.f};

#pragma unroll
  for (int j = 0; j < NTW; ++j) {
    int n = ((wave + 8*j) << 4) + lh;
    __builtin_prefetch(WT + (size_t)n * Kpad, 0, 0);   // global_prefetch_b8
  }

  for (int k0 = 0; k0 < Kpad; k0 += 32) {
    // A 16-bit 16x32 layout: lanes0-15 K=k0+0..7 & k0+16..23 ; lanes16-31 +8
    ABfrag a[4];
#pragma unroll
    for (int mt = 0; mt < 4; ++mt) {
      const _Float16* ap = X + ((mt << 4) + lh) * ldx + k0 + (lg << 3);
      a[mt].u[0] = *(const u32x4*)(ap);
      a[mt].u[1] = *(const u32x4*)(ap + 16);
    }
#pragma unroll
    for (int j = 0; j < NTW; ++j) {
      int n = ((wave + 8*j) << 4) + lh;
      // B 16-bit 32x16 layout: lanes0-15 K=k0+0..7 & k0+16..23 ; lanes16-31 +8
      const _Float16* bp = WT + (size_t)n * Kpad + k0 + (lg << 3);
      ABfrag b;
      b.u[0] = *(const u32x4*)(bp);
      b.u[1] = *(const u32x4*)(bp + 16);
#pragma unroll
      for (int mt = 0; mt < 4; ++mt)
        acc[j][mt] = __builtin_amdgcn_wmma_f32_16x16x32_f16(false, a[mt].v, false, b.v,
                                                            (short)0, acc[j][mt],
                                                            false, false);
    }
  }

  // epilogue: D vgpr r -> row r (lanes<16) / row r+8 (lanes>=16), col = lane&15
#pragma unroll
  for (int j = 0; j < NTW; ++j) {
    int n = ((wave + 8*j) << 4) + lh;
    float bv = bias[n];
#pragma unroll
    for (int mt = 0; mt < 4; ++mt) {
      int mbase = (mt << 4) + (lg << 3);
#pragma unroll
      for (int r = 0; r < 8; ++r) {
        float v = acc[j][mt][r] + bv;
        if (do_relu) v = fmaxf(v, 0.f);
        Y[(mbase + r) * ldy + n] = (_Float16)v;
      }
    }
  }
}

// ---------- fused NeRF MLP: posenc -> 8 layers(+skip) -> sigma/feat/branch/rgb ----------
__global__ __launch_bounds__(256) void nerf_mlp_kernel(
    const float* __restrict__ ro, const float* __restrict__ rd,
    const float* __restrict__ tvals, int NS, int M,
    MlpWeights w, float* __restrict__ sigma_out, float* __restrict__ rgb_out)
{
  __shared__ _Float16 Xa[MT * LDA];   // 32 KB
  __shared__ _Float16 Xb[MT * LDB];   // 40 KB

  int tid = threadIdx.x;
  int p0  = blockIdx.x * MT;

  float px = 0.f, py = 0.f, pz = 0.f, dx = 0.f, dy = 0.f, dz = 0.f;
  bool active = (tid < MT) && (p0 + tid < M);
  if (active) {
    int p = p0 + tid;
    int ray = p / NS;
    float t = tvals[p];
    dx = rd[ray*3+0]; dy = rd[ray*3+1]; dz = rd[ray*3+2];
    px = ro[ray*3+0] + dx * t;
    py = ro[ray*3+1] + dy * t;
    pz = ro[ray*3+2] + dz * t;
    write_enc(&Xa[tid * LDA], px, py, pz);
  }
  __syncthreads();

  gemm_layer<2>(Xa, LDA,  64, w.W0T, w.b0, Xb, LDB, true);  __syncthreads();
  gemm_layer<2>(Xb, LDB, 256, w.W1T, w.b1, Xa, LDA, true);  __syncthreads();
  gemm_layer<2>(Xa, LDA, 256, w.W2T, w.b2, Xb, LDB, true);  __syncthreads();
  gemm_layer<2>(Xb, LDB, 256, w.W3T, w.b3, Xa, LDA, true);  __syncthreads();
  gemm_layer<2>(Xa, LDA, 256, w.W4T, w.b4, Xb, LDB, true);
  // skip concat: Xb cols 256..319 = posenc(pts)  (disjoint cols from gemm output)
  if (active) write_enc(&Xb[tid * LDB + 256], px, py, pz);
  __syncthreads();
  gemm_layer<2>(Xb, LDB, 320, w.W5T, w.b5, Xa, LDA, true);  __syncthreads();
  gemm_layer<2>(Xa, LDA, 256, w.W6T, w.b6, Xb, LDB, true);  __syncthreads();
  gemm_layer<2>(Xb, LDB, 256, w.W7T, w.b7, Xa, LDA, true);  __syncthreads();  // H7 in Xa

  // sigma head (reads Xa; feat gemm also only reads Xa -> safe concurrent)
  if (active) {
    float s = w.sigb[0];
    const _Float16* h = &Xa[tid * LDA];
    for (int k = 0; k < 256; ++k) s += (float)h[k] * w.sigw[k];
    sigma_out[p0 + tid] = fmaxf(s, 0.f);
  }
  gemm_layer<2>(Xa, LDA, 256, w.featT, w.featb, Xb, LDB, false);
  // branch concat: Xb cols 256..319 = posenc(dirs)
  if (active) write_enc(&Xb[tid * LDB + 256], dx, dy, dz);
  __syncthreads();
  gemm_layer<1>(Xb, LDB, 320, w.branchT, w.branchb, Xa, LDA, true);
  __syncthreads();

  // rgb head: 3 dots of 128 + sigmoid
  if (tid < MT * 3) {
    int mp = tid / 3, ch = tid - mp * 3;
    int p = p0 + mp;
    if (p < M) {
      float a = w.rgbb[ch];
      const _Float16* h = &Xa[mp * LDA];
      for (int k = 0; k < 128; ++k) a += (float)h[k] * w.rgbw[k * 3 + ch];
      rgb_out[(size_t)p * 3 + ch] = 1.f / (1.f + __expf(-a));
    }
  }
}

// ---------- volume rendering (one thread per ray) ----------
__global__ void render_kernel(const float* __restrict__ tvals,
                              const float* __restrict__ sigma,
                              const float* __restrict__ rgb, int NS,
                              float* __restrict__ image, float* __restrict__ depth,
                              float* __restrict__ wout)
{
  int r = blockIdx.x * blockDim.x + threadIdx.x;
  if (r >= 1024) return;
  const float* t = tvals + (size_t)r * NS;
  float T = 1.f, i0 = 0.f, i1 = 0.f, i2 = 0.f, dep = 0.f;
  for (int i = 0; i < NS; ++i) {
    float dist = (i + 1 < NS) ? (t[i+1] - t[i]) : 1e10f;
    float a = 1.f - __expf(-sigma[(size_t)r * NS + i] * dist);
    float wgt = a * T;
    const float* c = rgb + ((size_t)r * NS + i) * 3;
    i0 += wgt * c[0]; i1 += wgt * c[1]; i2 += wgt * c[2];
    dep += wgt * t[i];
    if (wout) wout[(size_t)r * NS + i] = wgt;
    T *= (1.f - a + 1e-10f);
  }
  image[r*3+0] = i0; image[r*3+1] = i1; image[r*3+2] = i2;
  depth[r] = dep;
}

// ---------- inverse-CDF sampling + sorted merge (one block per ray) ----------
__global__ void sample_sort_kernel(const float* __restrict__ ct,
                                   const float* __restrict__ cw,
                                   const float* __restrict__ u,
                                   float* __restrict__ fine_t)
{
  int r = blockIdx.x, tid = threadIdx.x;
  __shared__ float t[64], cdfA[63], binsA[63], vals[192];
  if (tid < 64) t[tid] = ct[(size_t)r * 64 + tid];
  __syncthreads();
  if (tid < 63) binsA[tid] = 0.5f * (t[tid] + t[tid+1]);
  if (tid == 0) {
    float wv[62], S = 0.f;
    for (int i = 0; i < 62; ++i) { wv[i] = cw[(size_t)r*64 + 1 + i] + 1e-10f; S += wv[i]; }
    float c = 0.f; cdfA[0] = 0.f;
    for (int i = 0; i < 62; ++i) { c += wv[i] / S; cdfA[i+1] = c; }
  }
  __syncthreads();
  if (tid < 64) vals[tid] = t[tid];
  if (tid < 128) {
    float uu = u[(size_t)r * 128 + tid];
    int ind = 0;
    for (int j = 0; j < 63; ++j) ind += (cdfA[j] <= uu) ? 1 : 0;   // searchsorted right
    int below = ind - 1 > 0 ? ind - 1 : 0;
    int above = ind < 62 ? ind : 62;
    float cb = cdfA[below], ca = cdfA[above];
    float bb = binsA[below], ba = binsA[above];
    float den = (ca - cb < 1e-5f) ? 1.f : (ca - cb);
    vals[64 + tid] = bb + (uu - cb) / den * (ba - bb);
  }
  __syncthreads();
  if (tid < 192) {
    float v = vals[tid]; int rank = 0;
    for (int j = 0; j < 192; ++j) {
      float o = vals[j];
      rank += (o < v || (o == v && j < tid)) ? 1 : 0;
    }
    fine_t[(size_t)r * 192 + rank] = v;
  }
}

// ---------- host side ----------
struct ParamSet {
  const float *W[8], *b[8], *featw, *featb, *branchw, *branchb, *rgbw, *rgbb, *sigw, *sigb;
};

static ParamSet get_set(void* const* d_in, const int* in_sizes, int base) {
  ParamSet s;
  bool alpha = (in_sizes[base] == 128);   // branch_b first => alphabetical dict order
  if (alpha) {
    s.branchb = (const float*)d_in[base+0];
    s.branchw = (const float*)d_in[base+1];
    s.featb   = (const float*)d_in[base+2];
    s.featw   = (const float*)d_in[base+3];
    for (int i = 0; i < 8; ++i) {
      s.W[i] = (const float*)d_in[base+4+2*i];
      s.b[i] = (const float*)d_in[base+5+2*i];
    }
    s.rgbb = (const float*)d_in[base+20];
    s.rgbw = (const float*)d_in[base+21];
    s.sigb = (const float*)d_in[base+22];
    s.sigw = (const float*)d_in[base+23];
  } else {                                 // insertion order
    for (int i = 0; i < 8; ++i) {
      s.W[i] = (const float*)d_in[base+2*i];
      s.b[i] = (const float*)d_in[base+1+2*i];
    }
    s.sigw    = (const float*)d_in[base+16];
    s.sigb    = (const float*)d_in[base+17];
    s.featw   = (const float*)d_in[base+18];
    s.featb   = (const float*)d_in[base+19];
    s.branchw = (const float*)d_in[base+20];
    s.branchb = (const float*)d_in[base+21];
    s.rgbw    = (const float*)d_in[base+22];
    s.rgbb    = (const float*)d_in[base+23];
  }
  return s;
}

extern "C" void kernel_launch(void* const* d_in, const int* in_sizes, int n_in,
                              void* d_out, int out_size, void* d_ws, size_t ws_size,
                              hipStream_t stream) {
  (void)n_in; (void)out_size; (void)ws_size;
  const float* ro = (const float*)d_in[0];
  const float* rd = (const float*)d_in[1];
  const float* ct = (const float*)d_in[2];
  const float* u  = (const float*)d_in[3];
  ParamSet pc = get_set(d_in, in_sizes, 4);
  ParamSet pf = get_set(d_in, in_sizes, 28);

  // workspace layout
  char* ws = (char*)d_ws;
  size_t off = 0;
  auto walloc = [&](size_t bytes) -> void* {
    void* p = ws + off;
    off += (bytes + 255) & ~(size_t)255;
    return p;
  };
  const size_t SET_E = 598016;               // f16 elems per packed weight set
  _Float16* wts      = (_Float16*)walloc(2 * SET_E * sizeof(_Float16));
  float* sigma_buf   = (float*)walloc(196608 * sizeof(float));
  float* rgb_buf     = (float*)walloc(196608 * 3 * sizeof(float));
  float* cw_buf      = (float*)walloc(1024 * 64 * sizeof(float));
  float* fine_t      = (float*)walloc(1024 * 192 * sizeof(float));

  // per-set packed offsets (f16 elems)
  const size_t oW0 = 0, oW1 = 16384, oW2 = oW1 + 65536, oW3 = oW2 + 65536,
               oW4 = oW3 + 65536, oW5 = oW4 + 65536, oW6 = oW5 + 81920,
               oW7 = oW6 + 65536, oFe = oW7 + 65536, oBr = oFe + 65536;

  auto pack = [&](const float* W, _Float16* dst, int K, int N, int Kpad) {
    int total = N * Kpad;
    pack_wt_kernel<<<(total + 255) / 256, 256, 0, stream>>>(W, dst, K, N, Kpad);
  };

  MlpWeights mw[2];
  const ParamSet* ps[2] = {&pc, &pf};
  for (int s = 0; s < 2; ++s) {
    _Float16* base = wts + s * SET_E;
    const ParamSet& P = *ps[s];
    pack(P.W[0], base + oW0,  63, 256,  64);
    pack(P.W[1], base + oW1, 256, 256, 256);
    pack(P.W[2], base + oW2, 256, 256, 256);
    pack(P.W[3], base + oW3, 256, 256, 256);
    pack(P.W[4], base + oW4, 256, 256, 256);
    pack(P.W[5], base + oW5, 319, 256, 320);
    pack(P.W[6], base + oW6, 256, 256, 256);
    pack(P.W[7], base + oW7, 256, 256, 256);
    pack(P.featw,   base + oFe, 256, 256, 256);
    pack(P.branchw, base + oBr, 319, 128, 320);
    MlpWeights& w = mw[s];
    w.W0T = base + oW0; w.W1T = base + oW1; w.W2T = base + oW2; w.W3T = base + oW3;
    w.W4T = base + oW4; w.W5T = base + oW5; w.W6T = base + oW6; w.W7T = base + oW7;
    w.featT = base + oFe; w.branchT = base + oBr;
    w.b0 = P.b[0]; w.b1 = P.b[1]; w.b2 = P.b[2]; w.b3 = P.b[3];
    w.b4 = P.b[4]; w.b5 = P.b[5]; w.b6 = P.b[6]; w.b7 = P.b[7];
    w.featb = P.featb; w.branchb = P.branchb;
    w.sigw = P.sigw; w.sigb = P.sigb; w.rgbw = P.rgbw; w.rgbb = P.rgbb;
  }

  float* out        = (float*)d_out;
  float* out_img_c  = out;           // 1024*3
  float* out_dep_c  = out + 3072;    // 1024
  float* out_img_f  = out + 4096;    // 1024*3
  float* out_dep_f  = out + 7168;    // 1024

  // ---- coarse pass ----
  nerf_mlp_kernel<<<65536 / MT, 256, 0, stream>>>(ro, rd, ct, 64, 65536,
                                                  mw[0], sigma_buf, rgb_buf);
  render_kernel<<<4, 256, 0, stream>>>(ct, sigma_buf, rgb_buf, 64,
                                       out_img_c, out_dep_c, cw_buf);
  // ---- hierarchical resampling + merge sort ----
  sample_sort_kernel<<<1024, 192, 0, stream>>>(ct, cw_buf, u, fine_t);
  // ---- fine pass ----
  nerf_mlp_kernel<<<196608 / MT, 256, 0, stream>>>(ro, rd, fine_t, 192, 196608,
                                                   mw[1], sigma_buf, rgb_buf);
  render_kernel<<<4, 256, 0, stream>>>(fine_t, sigma_buf, rgb_buf, 192,
                                       out_img_f, out_dep_f, nullptr);
}
